// NeuralCDELSTM_53180285059415
// MI455X (gfx1250) — compile-verified, matrix operand order checked by reading, not run on hardware
//
#include <hip/hip_runtime.h>

// ---- problem constants (match reference) ----
#define B_      4096
#define HID_    128
#define INCH_   16
#define LKNOT_  32
#define LSTM_   256
#define OUT_    15
#define NSTEPS_ 149
#define G4N_    2048   // IN_CH * HID

typedef __attribute__((ext_vector_type(8)))  float   v8f;
typedef __attribute__((ext_vector_type(16))) __bf16  v16bf;
typedef __attribute__((ext_vector_type(8)))  __bf16  v8bf;

// Load one WMMA bf16 operand fragment for this lane.
// ISA layout (16-bit A 16x32): lane idx holds row, elements 0..7 = K[8*hi .. 8*hi+7],
// elements 8..15 = K[16+8*hi .. 16+8*hi+7]  -> two contiguous 16-byte chunks.
__device__ __forceinline__ v16bf load_frag(const __bf16* p) {
    v8bf lo = *reinterpret_cast<const v8bf*>(p);
    v8bf hi = *reinterpret_cast<const v8bf*>(p + 16);
    return __builtin_shufflevector(lo, hi, 0,1,2,3,4,5,6,7,8,9,10,11,12,13,14,15);
}

#define WMMA_BF16(a, b, c) \
    __builtin_amdgcn_wmma_f32_16x16x32_bf16(false, (a), false, (b), (short)0, (c), false, false)

// ---------------------------------------------------------------------------
// 2x2 register-blocked GEMM: C = act(A(MxK bf16,lda) @ W(NxK bf16)^T + bias)
// 8 waves/block arranged 4(M) x 2(N); each wave computes a 32x32 tile with
// four 16x16 WMMA accumulators, reusing each fragment twice (~32 flop/byte).
// Block tile = 128(M) x 64(N); grid = (M/128, N/64). Requires N % 64 == 0.
// Dual store: f32 (ldoF) and/or bf16 (ldoB).
// ---------------------------------------------------------------------------
__global__ void gemm_wmma2_k(const __bf16* __restrict__ A, int lda,
                             const __bf16* __restrict__ W,
                             const float*  __restrict__ bias,
                             float* __restrict__ outF, long long ldoF,
                             __bf16* __restrict__ outB, int ldoB,
                             int K, int act)
{
    int wave = threadIdx.x >> 5;
    int lane = threadIdx.x & 31;
    int idx  = lane & 15;
    int hi   = lane >> 4;
    int wm   = wave & 3;          // 4 waves along M
    int wn   = wave >> 2;         // 2 waves along N
    int row0 = (blockIdx.x * 4 + wm) * 32;
    int col0 = (blockIdx.y * 2 + wn) * 32;

    const __bf16* ap0 = A + (size_t)(row0 + idx) * lda + 8 * hi;
    const __bf16* ap1 = ap0 + (size_t)16 * lda;
    const __bf16* wp0 = W + (size_t)(col0 + idx) * K + 8 * hi;
    const __bf16* wp1 = wp0 + (size_t)16 * K;

    v8f acc00 = {0.f,0.f,0.f,0.f,0.f,0.f,0.f,0.f};
    v8f acc01 = acc00, acc10 = acc00, acc11 = acc00;
    for (int k0 = 0; k0 < K; k0 += 32) {
        v16bf a0 = load_frag(ap0 + k0);
        v16bf a1 = load_frag(ap1 + k0);
        v16bf b0 = load_frag(wp0 + k0);
        v16bf b1 = load_frag(wp1 + k0);
        acc00 = WMMA_BF16(a0, b0, acc00);
        acc01 = WMMA_BF16(a0, b1, acc01);
        acc10 = WMMA_BF16(a1, b0, acc10);
        acc11 = WMMA_BF16(a1, b1, acc11);
    }

    float bv0 = bias[col0 + idx];
    float bv1 = bias[col0 + 16 + idx];
    #pragma unroll
    for (int mi = 0; mi < 2; ++mi) {
        const v8f* aN[2] = { mi ? &acc10 : &acc00, mi ? &acc11 : &acc01 };
        #pragma unroll
        for (int r = 0; r < 8; ++r) {
            int row = row0 + 16 * mi + hi * 8 + r;
            float v0 = (*aN[0])[r] + bv0;
            float v1 = (*aN[1])[r] + bv1;
            if (act) { v0 = fmaxf(v0, 0.f); v1 = fmaxf(v1, 0.f); }
            if (outF) {
                outF[(size_t)row * ldoF + col0 + idx]      = v0;
                outF[(size_t)row * ldoF + col0 + 16 + idx] = v1;
            }
            if (outB) {
                outB[(size_t)row * ldoB + col0 + idx]      = (__bf16)v0;
                outB[(size_t)row * ldoB + col0 + 16 + idx] = (__bf16)v1;
            }
        }
    }
}

// ---------------------------------------------------------------------------
// 1x1 GEMM with column masking (only used for the tiny fc: N=16, nValid=15).
// ---------------------------------------------------------------------------
__global__ void gemm_wmma_k(const __bf16* __restrict__ A, int lda,
                            const __bf16* __restrict__ W,
                            const float*  __restrict__ bias,
                            float* __restrict__ outF, long long ldoF,
                            __bf16* __restrict__ outB, int ldoB,
                            int K, int nValid, int act)
{
    int wave = threadIdx.x >> 5;       // 8 waves along M, 1 col tile
    int lane = threadIdx.x & 31;
    int idx  = lane & 15;
    int hi   = lane >> 4;
    int row0 = (blockIdx.x * 8 + wave) * 16;
    int col0 = blockIdx.y * 16;

    const __bf16* ap = A + (size_t)(row0 + idx) * lda + 8 * hi;
    const __bf16* wp = W + (size_t)(col0 + idx) * K   + 8 * hi;

    v8f acc = {0.f,0.f,0.f,0.f,0.f,0.f,0.f,0.f};
    for (int k0 = 0; k0 < K; k0 += 32) {
        v16bf a = load_frag(ap + k0);
        v16bf b = load_frag(wp + k0);
        acc = WMMA_BF16(a, b, acc);
    }

    int col = col0 + idx;
    float bv = bias[col];
    if (col < nValid) {
        #pragma unroll
        for (int r = 0; r < 8; ++r) {
            float v = acc[r] + bv;
            if (act) v = fmaxf(v, 0.f);
            int row = row0 + hi * 8 + r;
            if (outF) outF[(size_t)row * ldoF + col] = v;
            if (outB) outB[(size_t)row * ldoB + col] = (__bf16)v;
        }
    }
}

// ---------------------------------------------------------------------------
// Fused last CDE layer, 2x2 blocked:
//   K[b,h] = sum_c tanh( (H3 @ W4^T + b4)[b, h*16+c] ) * dX[b,c]
// Each 16x16 WMMA tile == 16 batch rows x all 16 channels of one h, so the
// dX contraction is a 16-lane cross-lane reduction per accumulator.
// Wave tile: 32 batch rows x 2 hidden units. Block = 128 rows x 4 h.
// ---------------------------------------------------------------------------
__global__ void gemm4_einsum_k(const __bf16* __restrict__ H3,
                               const __bf16* __restrict__ W4,   // (2048,128)
                               const float*  __restrict__ b4,
                               const float*  __restrict__ coeffs, // (B, L, 16)
                               int step,
                               float* __restrict__ Kout)          // (B,128)
{
    int wave = threadIdx.x >> 5;
    int lane = threadIdx.x & 31;
    int idx  = lane & 15;
    int hi   = lane >> 4;
    int wm   = wave & 3;
    int wn   = wave >> 2;
    int row0 = (blockIdx.x * 4 + wm) * 32;
    int col0 = (blockIdx.y * 2 + wn) * 32;   // n = h*16 + c

    const __bf16* ap0 = H3 + (size_t)(row0 + idx) * HID_ + 8 * hi;
    const __bf16* ap1 = ap0 + (size_t)16 * HID_;
    const __bf16* wp0 = W4 + (size_t)(col0 + idx) * HID_ + 8 * hi;
    const __bf16* wp1 = wp0 + (size_t)16 * HID_;

    v8f acc00 = {0.f,0.f,0.f,0.f,0.f,0.f,0.f,0.f};
    v8f acc01 = acc00, acc10 = acc00, acc11 = acc00;
    #pragma unroll
    for (int k0 = 0; k0 < HID_; k0 += 32) {
        v16bf a0 = load_frag(ap0 + k0);
        v16bf a1 = load_frag(ap1 + k0);
        v16bf b0 = load_frag(wp0 + k0);
        v16bf b1 = load_frag(wp1 + k0);
        acc00 = WMMA_BF16(a0, b0, acc00);
        acc01 = WMMA_BF16(a0, b1, acc01);
        acc10 = WMMA_BF16(a1, b0, acc10);
        acc11 = WMMA_BF16(a1, b1, acc11);
    }

    float bv0 = b4[col0 + idx];
    float bv1 = b4[col0 + 16 + idx];
    float vals[2][2][8];
    #pragma unroll
    for (int mi = 0; mi < 2; ++mi) {
        const v8f* aN[2] = { mi ? &acc10 : &acc00, mi ? &acc11 : &acc01 };
        #pragma unroll
        for (int r = 0; r < 8; ++r) {
            int row = row0 + 16 * mi + hi * 8 + r;
            const float* cp = coeffs + (size_t)row * (LKNOT_ * INCH_) + step * INCH_ + idx;
            float dx = cp[INCH_] - cp[0];        // coeffs[:,s+1]-coeffs[:,s]
            vals[mi][0][r] = tanhf((*aN[0])[r] + bv0) * dx;
            vals[mi][1][r] = tanhf((*aN[1])[r] + bv1) * dx;
        }
    }
    // reduce over the 16 channel columns (lanes with same hi)
    #pragma unroll
    for (int m = 1; m < 16; m <<= 1) {
        #pragma unroll
        for (int mi = 0; mi < 2; ++mi)
            #pragma unroll
            for (int ni = 0; ni < 2; ++ni)
                #pragma unroll
                for (int r = 0; r < 8; ++r)
                    vals[mi][ni][r] += __shfl_xor(vals[mi][ni][r], m, 32);
    }
    if (idx == 0) {
        int h = col0 >> 4;
        #pragma unroll
        for (int mi = 0; mi < 2; ++mi)
            #pragma unroll
            for (int ni = 0; ni < 2; ++ni)
                #pragma unroll
                for (int r = 0; r < 8; ++r)
                    Kout[(size_t)(row0 + 16 * mi + hi * 8 + r) * HID_ + (h + ni)] =
                        vals[mi][ni][r];
    }
}

// RK4 bookkeeping on (B,HID)
__global__ void rk4_update_k(float* __restrict__ z, float* __restrict__ ksum,
                             const float* __restrict__ k,
                             __bf16* __restrict__ zst, __bf16* __restrict__ zbf,
                             int phase)
{
    int i = blockIdx.x * blockDim.x + threadIdx.x;
    if (i >= B_ * HID_) return;
    float kv = k[i];
    if (phase == 0)      { ksum[i] = kv;            zst[i] = (__bf16)(z[i] + 0.5f * kv); }
    else if (phase == 1) { ksum[i] += 2.f * kv;     zst[i] = (__bf16)(z[i] + 0.5f * kv); }
    else if (phase == 2) { ksum[i] += 2.f * kv;     zst[i] = (__bf16)(z[i] + kv); }
    else { float zn = z[i] + (ksum[i] + kv) * (1.f / 6.f); z[i] = zn; zbf[i] = (__bf16)zn; }
}

// LSTM cell pointwise: gates (B,1024) f32 (i,f,g,o), c in/out f32, h -> 1 or 2 bf16 slots
__global__ void lstm_cell_k(const float* __restrict__ gates, float* __restrict__ c,
                            __bf16* __restrict__ o1, int ld1,
                            __bf16* __restrict__ o2, int ld2)
{
    int i = blockIdx.x * blockDim.x + threadIdx.x;
    if (i >= B_ * LSTM_) return;
    int b = i / LSTM_, n = i % LSTM_;
    const float* g = gates + (size_t)b * (4 * LSTM_);
    float ig = 1.f / (1.f + expf(-g[n]));
    float fg = 1.f / (1.f + expf(-g[LSTM_ + n]));
    float gg = tanhf(g[2 * LSTM_ + n]);
    float og = 1.f / (1.f + expf(-g[3 * LSTM_ + n]));
    float cn = fg * c[i] + ig * gg;
    c[i] = cn;
    float h = og * tanhf(cn);
    o1[(size_t)b * ld1 + n] = (__bf16)h;
    if (o2) o2[(size_t)b * ld2 + n] = (__bf16)h;
}

// h0 -> ha slot of cat0A, hb slot of cat1A, c1, c2 ; zero x-pad slot of cat0A
__global__ void lstm_init_k(const float* __restrict__ h0,
                            __bf16* __restrict__ cat0A, __bf16* __restrict__ cat1A,
                            float* __restrict__ c1, float* __restrict__ c2)
{
    int i = blockIdx.x * blockDim.x + threadIdx.x;
    if (i >= B_ * LSTM_) return;
    int b = i / LSTM_, n = i % LSTM_;
    float v = h0[i];
    cat0A[(size_t)b * 288 + 32 + n]  = (__bf16)v;
    cat1A[(size_t)b * 512 + 256 + n] = (__bf16)v;
    c1[i] = v;
    c2[i] = v;
    if (n < 32) cat0A[(size_t)b * 288 + n] = (__bf16)0.f;   // zero-padded x input
}

// ---- small prep kernels ----
__global__ void zero_bf16_k(__bf16* p, int n) {
    int i = blockIdx.x * blockDim.x + threadIdx.x;
    if (i < n) p[i] = (__bf16)0.f;
}
__global__ void cvt_k(__bf16* d, const float* s, int n) {
    int i = blockIdx.x * blockDim.x + threadIdx.x;
    if (i < n) d[i] = (__bf16)s[i];
}
__global__ void copy2d_k(__bf16* d, int ldd, const float* s, int lds, int rows, int cols) {
    int i = blockIdx.x * blockDim.x + threadIdx.x;
    if (i >= rows * cols) return;
    int r = i / cols, c = i % cols;
    d[(size_t)r * ldd + c] = (__bf16)s[(size_t)r * lds + c];
}
__global__ void add_f32_k(float* d, const float* a, const float* b, int n) {
    int i = blockIdx.x * blockDim.x + threadIdx.x;
    if (i < n) d[i] = a[i] + b[i];
}
__global__ void pad_f32_k(float* d, const float* s, int n, int total) {
    int i = blockIdx.x * blockDim.x + threadIdx.x;
    if (i < total) d[i] = (i < n) ? s[i] : 0.f;
}

extern "C" void kernel_launch(void* const* d_in, const int* in_sizes, int n_in,
                              void* d_out, int out_size, void* d_ws, size_t ws_size,
                              hipStream_t stream) {
    (void)in_sizes; (void)n_in; (void)out_size; (void)ws_size;
    const float* coeffs = (const float*)d_in[0];
    const float* W1 = (const float*)d_in[1];  const float* b1 = (const float*)d_in[2];
    const float* W2 = (const float*)d_in[3];  const float* b2 = (const float*)d_in[4];
    const float* W3 = (const float*)d_in[5];  const float* b3 = (const float*)d_in[6];
    const float* W4 = (const float*)d_in[7];  const float* b4 = (const float*)d_in[8];
    const float* Wi = (const float*)d_in[9];  const float* bi = (const float*)d_in[10];
    const float* Wr = (const float*)d_in[11]; const float* br = (const float*)d_in[12];
    const float* Wih0 = (const float*)d_in[13]; const float* Whh0 = (const float*)d_in[14];
    const float* bih0 = (const float*)d_in[15]; const float* bhh0 = (const float*)d_in[16];
    const float* Wih1 = (const float*)d_in[17]; const float* Whh1 = (const float*)d_in[18];
    const float* bih1 = (const float*)d_in[19]; const float* bhh1 = (const float*)d_in[20];
    const float* Wf = (const float*)d_in[21]; const float* bf = (const float*)d_in[22];
    float* out = (float*)d_out;

    // ---- carve workspace (256B aligned) ----
    char* wp = (char*)d_ws;
    auto alloc = [&](size_t bytes) -> void* {
        void* r = (void*)wp; wp += (bytes + 255) & ~(size_t)255; return r;
    };
    __bf16* W1b   = (__bf16*)alloc(16384 * 2);
    __bf16* W2b   = (__bf16*)alloc(16384 * 2);
    __bf16* W3b   = (__bf16*)alloc(16384 * 2);
    __bf16* W4b   = (__bf16*)alloc((size_t)G4N_ * HID_ * 2);
    __bf16* Wib   = (__bf16*)alloc(128 * 32 * 2);
    __bf16* Wrb   = (__bf16*)alloc(256 * 128 * 2);
    __bf16* cat0W = (__bf16*)alloc((size_t)1024 * 288 * 2);
    __bf16* cat1W = (__bf16*)alloc((size_t)1024 * 512 * 2);
    __bf16* Wfb   = (__bf16*)alloc(16 * 256 * 2);
    float*  bcat0 = (float*)alloc(1024 * 4);
    float*  bcat1 = (float*)alloc(1024 * 4);
    float*  bfp   = (float*)alloc(16 * 4);
    __bf16* c0pad = (__bf16*)alloc((size_t)B_ * 32 * 2);
    float*  z     = (float*)alloc((size_t)B_ * HID_ * 4);
    __bf16* zbf   = (__bf16*)alloc((size_t)B_ * HID_ * 2);
    __bf16* zstbf = (__bf16*)alloc((size_t)B_ * HID_ * 2);
    __bf16* h1b   = (__bf16*)alloc((size_t)B_ * HID_ * 2);
    __bf16* h2b   = (__bf16*)alloc((size_t)B_ * HID_ * 2);
    __bf16* h3b   = (__bf16*)alloc((size_t)B_ * HID_ * 2);
    float*  kbuf  = (float*)alloc((size_t)B_ * HID_ * 4);
    float*  ksum  = (float*)alloc((size_t)B_ * HID_ * 4);
    float*  h0tmp = (float*)alloc((size_t)B_ * LSTM_ * 4);
    __bf16* cat0A = (__bf16*)alloc((size_t)B_ * 288 * 2);
    __bf16* cat1A = (__bf16*)alloc((size_t)B_ * 512 * 2);
    float*  gates = (float*)alloc((size_t)B_ * 1024 * 4);
    float*  c1    = (float*)alloc((size_t)B_ * LSTM_ * 4);
    float*  c2    = (float*)alloc((size_t)B_ * LSTM_ * 4);

    auto g1 = [](int n) { return dim3((unsigned)((n + 255) / 256)); };
    // 2x2 blocked GEMM: block tile 128(M) x 64(N); requires N % 64 == 0.
    auto gemm = [&](const __bf16* A, int lda, const __bf16* W, const float* bias,
                    float* oF, long long ldoF, __bf16* oB, int ldoB,
                    int M, int N, int K, int act) {
        dim3 grid((unsigned)(M / 128), (unsigned)(N / 64));
        gemm_wmma2_k<<<grid, 256, 0, stream>>>(A, lda, W, bias, oF, ldoF, oB, ldoB, K, act);
    };

    // ---- weight prep (bf16 convert / pad / concat) ----
    cvt_k<<<g1(16384), 256, 0, stream>>>(W1b, W1, 16384);
    cvt_k<<<g1(16384), 256, 0, stream>>>(W2b, W2, 16384);
    cvt_k<<<g1(16384), 256, 0, stream>>>(W3b, W3, 16384);
    cvt_k<<<g1(G4N_ * HID_), 256, 0, stream>>>(W4b, W4, G4N_ * HID_);
    cvt_k<<<g1(256 * 128), 256, 0, stream>>>(Wrb, Wr, 256 * 128);
    zero_bf16_k<<<g1(128 * 32), 256, 0, stream>>>(Wib, 128 * 32);
    copy2d_k<<<g1(128 * 16), 256, 0, stream>>>(Wib, 32, Wi, 16, 128, 16);
    zero_bf16_k<<<g1(1024 * 288), 256, 0, stream>>>(cat0W, 1024 * 288);
    copy2d_k<<<g1(1024 * 15), 256, 0, stream>>>(cat0W, 288, Wih0, 15, 1024, 15);
    copy2d_k<<<g1(1024 * 256), 256, 0, stream>>>(cat0W + 32, 288, Whh0, 256, 1024, 256);
    copy2d_k<<<g1(1024 * 256), 256, 0, stream>>>(cat1W, 512, Wih1, 256, 1024, 256);
    copy2d_k<<<g1(1024 * 256), 256, 0, stream>>>(cat1W + 256, 512, Whh1, 256, 1024, 256);
    zero_bf16_k<<<g1(16 * 256), 256, 0, stream>>>(Wfb, 16 * 256);
    copy2d_k<<<g1(15 * 256), 256, 0, stream>>>(Wfb, 256, Wf, 256, 15, 256);
    add_f32_k<<<g1(1024), 256, 0, stream>>>(bcat0, bih0, bhh0, 1024);
    add_f32_k<<<g1(1024), 256, 0, stream>>>(bcat1, bih1, bhh1, 1024);
    pad_f32_k<<<g1(16), 32, 0, stream>>>(bfp, bf, 15, 16);
    zero_bf16_k<<<g1(B_ * 32), 256, 0, stream>>>(c0pad, B_ * 32);
    copy2d_k<<<g1(B_ * 16), 256, 0, stream>>>(c0pad, 32, coeffs, LKNOT_ * INCH_, B_, 16);

    // ---- z0 = coeffs[:,0] @ Wi^T + bi ----
    gemm(c0pad, 32, Wib, bi, z, HID_, zbf, HID_, B_, HID_, 32, 0);

    // ---- CDE: 31 RK4 steps ----
    for (int s = 0; s < LKNOT_ - 1; ++s) {
        for (int e = 0; e < 4; ++e) {
            const __bf16* zin = (e == 0) ? zbf : zstbf;
            gemm(zin, HID_, W1b, b1, nullptr, 0, h1b, HID_, B_, HID_, HID_, 1);
            gemm(h1b, HID_, W2b, b2, nullptr, 0, h2b, HID_, B_, HID_, HID_, 1);
            gemm(h2b, HID_, W3b, b3, nullptr, 0, h3b, HID_, B_, HID_, HID_, 1);
            gemm4_einsum_k<<<dim3(B_ / 128, G4N_ / 64), 256, 0, stream>>>(
                h3b, W4b, b4, coeffs, s, kbuf);
            rk4_update_k<<<g1(B_ * HID_), 256, 0, stream>>>(z, ksum, kbuf, zstbf, zbf, e);
        }
    }

    // ---- readout h0 = zT @ Wr^T + br ; init LSTM state ----
    gemm(zbf, HID_, Wrb, br, h0tmp, LSTM_, nullptr, 0, B_, LSTM_, HID_, 0);
    lstm_init_k<<<g1(B_ * LSTM_), 256, 0, stream>>>(h0tmp, cat0A, cat1A, c1, c2);

    // ---- autoregressive decode: 149 steps ----
    for (int t = 0; t < NSTEPS_; ++t) {
        // layer 0: gates = [x|ha] @ [Wih0|Whh0]^T + (bih0+bhh0)
        gemm(cat0A, 288, cat0W, bcat0, gates, 1024, nullptr, 0, B_, 1024, 288, 0);
        // cell0 -> h1 into cat1A[:,0:256) (layer1 input) and cat0A[:,32:288) (next ha)
        lstm_cell_k<<<g1(B_ * LSTM_), 256, 0, stream>>>(gates, c1, cat1A, 512, cat0A + 32, 288);
        // layer 1: gates = [h1|hb] @ [Wih1|Whh1]^T + (bih1+bhh1)
        gemm(cat1A, 512, cat1W, bcat1, gates, 1024, nullptr, 0, B_, 1024, 512, 0);
        // cell1 -> h2 into cat1A[:,256:512) (next hb, also fc input)
        lstm_cell_k<<<g1(B_ * LSTM_), 256, 0, stream>>>(gates, c2, cat1A + 256, 512, nullptr, 0);
        // fc: out[:,t,:] = h2 @ Wf^T + bf ; also feed back as next x (bf16, cols<15)
        gemm_wmma_k<<<dim3(B_ / 128, 1), 256, 0, stream>>>(
            cat1A + 256, 512, Wfb, bfp,
            out + (size_t)t * OUT_, (long long)NSTEPS_ * OUT_,
            cat0A, 288, 256, OUT_, 0);
    }
}
